// SelfAttnMatch_31799937860219
// MI455X (gfx1250) — compile-verified, matrix operand order checked by reading, not run on hardware
//
#include <hip/hip_runtime.h>

// B=16, L=1024, D=768
#define BATCH 16
#define SEQL  1024
#define DIM   768

typedef __attribute__((ext_vector_type(16))) __bf16 v16bf;
typedef __attribute__((ext_vector_type(8)))  float  v8f;
typedef unsigned int u32x4 __attribute__((ext_vector_type(4)));
typedef int          i32x4 __attribute__((ext_vector_type(4)));
typedef int          i32x8 __attribute__((ext_vector_type(8)));

union ABReg { v16bf v; unsigned u[8]; };

// pack two fp32 -> bf16x2 dword by truncation: single v_perm_b32
__device__ __forceinline__ unsigned pack2(float lo, float hi) {
#if __has_builtin(__builtin_amdgcn_perm)
  return __builtin_amdgcn_perm(__float_as_uint(hi), __float_as_uint(lo),
                               0x07060302u);
#else
  return (__float_as_uint(lo) >> 16) | (__float_as_uint(hi) & 0xffff0000u);
#endif
}
// fp32 -> bf16 truncation; folds into global_store_d16_hi_b16
__device__ __forceinline__ unsigned short f2bf1(float f) {
  return (unsigned short)(__float_as_uint(f) >> 16);
}
// K-pair base for VGPR v (0..7) and lane-half (ISA 7.12.2, 16-bit A/B 16x32 layout)
__device__ __forceinline__ int kpat(int v, int half) {
  return ((v & 4) << 2) + 8 * half + 2 * (v & 3);
}

// ---------------------------------------------------------------------------
// Kernel 0a: W fp32 -> bf16 (589824 elements, 2 per thread)
// ---------------------------------------------------------------------------
__global__ __launch_bounds__(256) void convw_kernel(
    const float* __restrict__ W, unsigned short* __restrict__ Wbf) {
  int i = blockIdx.x * 256 + threadIdx.x;           // dword index
  const float2 f = ((const float2*)W)[i];
  ((unsigned*)Wbf)[i] = pack2(f.x, f.y);
}

// ---------------------------------------------------------------------------
// Kernel 0b: x fp32 [B][L][D] -> bf16 transposed xT [B][D][L]
// 64x64 LDS tile transpose. grid = B * (L/64) * (D/64) = 3072 blocks.
// ---------------------------------------------------------------------------
__global__ __launch_bounds__(256) void transx_kernel(
    const float* __restrict__ x, unsigned short* __restrict__ xT) {
  __shared__ __align__(4) unsigned short tile[64][66];  // padded, 8.25 KB
  const int tid = threadIdx.x;
  const int blk = blockIdx.x;
  const int b  = blk / 192;
  const int lt = (blk % 192) / 12;
  const int dt = blk % 12;
  const int l0 = lt * 64, d0 = dt * 64;
  {
    const int c2 = (tid & 31) * 2;                  // d pair
    const int rb = tid >> 5;                        // l row group
#pragma unroll
    for (int i = 0; i < 8; ++i) {
      int r = rb + 8 * i;
      const float2 f =
          *(const float2*)&x[((size_t)b * SEQL + l0 + r) * DIM + d0 + c2];
      *(unsigned*)&tile[r][c2] = pack2(f.x, f.y);
    }
  }
  __syncthreads();
  {
    const int cc = (tid & 31) * 2;                  // l pair
    const int rb = tid >> 5;                        // d row group
#pragma unroll
    for (int i = 0; i < 8; ++i) {
      int rr = rb + 8 * i;
      unsigned u = (unsigned)tile[cc][rr] | ((unsigned)tile[cc + 1][rr] << 16);
      *(unsigned*)&xT[((size_t)b * DIM + d0 + rr) * SEQL + l0 + cc] = u;
    }
  }
}

// ---------------------------------------------------------------------------
// Kernel 1: x_proj = relu(x @ W^T + b) -> bf16. One wave per 16x64 tile
// (4 accumulators); A converted once per k-step, amortized over 4 WMMAs.
// Pointer-based addressing (round-3 codegen: s_clause + global_load_b128).
// ---------------------------------------------------------------------------
__global__ __launch_bounds__(256) void proj_kernel(
    const float* __restrict__ x, const unsigned short* __restrict__ Wbf,
    const float* __restrict__ bias, unsigned short* __restrict__ xproj) {
  const int lane = threadIdx.x & 31;
  const int wave = threadIdx.x >> 5;
  const int job  = blockIdx.x * 8 + wave;
  const int mtile = job / 12;
  const int nq    = job % 12;
  const int l15  = lane & 15;
  const int half = lane >> 4;

  const float2* xr = (const float2*)(x + (size_t)(mtile * 16 + l15) * DIM);
  const unsigned short* wr[4];
#pragma unroll
  for (int t = 0; t < 4; ++t)
    wr[t] = Wbf + (size_t)((nq * 4 + t) * 16 + l15) * DIM;

  v8f c0 = {}, c1 = {}, c2 = {}, c3 = {};
  for (int kb = 0; kb < DIM; kb += 32) {
    ABReg a;
#pragma unroll
    for (int v = 0; v < 8; ++v) {
      int k = kb + kpat(v, half);
      const float2 f = xr[k >> 1];
      a.u[v] = pack2(f.x, f.y);
    }
    ABReg b0, b1, b2, b3;
#pragma unroll
    for (int v = 0; v < 8; ++v) {
      int k = kb + kpat(v, half);
      b0.u[v] = *(const unsigned*)&wr[0][k];
      b1.u[v] = *(const unsigned*)&wr[1][k];
      b2.u[v] = *(const unsigned*)&wr[2][k];
      b3.u[v] = *(const unsigned*)&wr[3][k];
    }
    c0 = __builtin_amdgcn_wmma_f32_16x16x32_bf16(false, a.v, false, b0.v, (short)0, c0, false, false);
    c1 = __builtin_amdgcn_wmma_f32_16x16x32_bf16(false, a.v, false, b1.v, (short)0, c1, false, false);
    c2 = __builtin_amdgcn_wmma_f32_16x16x32_bf16(false, a.v, false, b2.v, (short)0, c2, false, false);
    c3 = __builtin_amdgcn_wmma_f32_16x16x32_bf16(false, a.v, false, b3.v, (short)0, c3, false, false);
  }
  v8f* cs[4] = {&c0, &c1, &c2, &c3};
#pragma unroll
  for (int t = 0; t < 4; ++t) {
    const int ncol = (nq * 4 + t) * 16 + l15;
    const float bv = bias[ncol];
#pragma unroll
    for (int r = 0; r < 8; ++r) {
      float val = (*cs[t])[r] + bv;
      val = val > 0.f ? val : 0.f;                 // relu
      int m = mtile * 16 + r + 8 * half;           // C/D row layout
      xproj[(size_t)m * DIM + ncol] = f2bf1(val);  // -> store_d16_hi
    }
  }
}

// ---------------------------------------------------------------------------
// Kernel 2: fused flash attention (Br=32, Bc=32).
// Q block staged by the Tensor Data Mover (TENSOR_LOAD_TO_LDS) when the
// builtin exists; manual copy fallback otherwise.
// ---------------------------------------------------------------------------
__global__ __launch_bounds__(256) void attn_kernel(
    const unsigned short* __restrict__ xT, const unsigned char* __restrict__ mask,
    const unsigned short* __restrict__ xproj, float* __restrict__ out) {
  __shared__ __align__(16) unsigned short s_Q[32 * DIM];   // 48 KB bf16 Q block
  __shared__ __align__(16) float          s_S0[32 * 32];   // 4 KB k-half 0
  __shared__ __align__(16) float          s_S1[32 * 32];   // 4 KB k-half 1
  __shared__ __align__(16) unsigned short s_P[32 * 32];    // 2 KB bf16 probs
  __shared__ float s_max[32], s_sum[32], s_scale[32];
  __shared__ unsigned char s_mask[32];

  const int tid  = threadIdx.x;
  const int lane = tid & 31;
  const int wave = tid >> 5;
  const int l15  = lane & 15;
  const int half = lane >> 4;
  const int b    = blockIdx.x >> 5;
  const int qb   = blockIdx.x & 31;
  const size_t brow0 = (size_t)b * SEQL;

  // ---- stage Qp (32 x 768 bf16) into LDS ----
#if __has_builtin(__builtin_amdgcn_tensor_load_to_lds)
  if (wave == 0) {
    // D# group 0: count=1 | lds_addr | global tile addr | type=2 ("image")
    const unsigned long long ga =
        (unsigned long long)(size_t)(const void*)(xproj +
            (brow0 + qb * 32) * DIM);
    u32x4 g0;
    g0.x = 1u;                                          // count=1, user mode
    g0.y = (unsigned)(size_t)(void*)s_Q;                // LDS byte address
    g0.z = (unsigned)(ga & 0xffffffffu);                // global_addr[31:0]
    g0.w = (unsigned)((ga >> 32) & 0x01ffffffu) | (2u << 30); // [56:32]|type=2
    // D# group 1: 2-byte elems, tensor 768 x 16384, tile 768 x 32, stride 768
    i32x8 g1;
    g1[0] = 0x00010000;                 // wg_mask=0, data_size=1 (2 bytes)
    g1[1] = (int)(768u << 16);          // tensor_dim0[15:0]=768 (bits 63:48)
    g1[2] = (int)(16384u << 16);        // dim0 hi=0 | tensor_dim1[15:0]=16384
    g1[3] = (int)(768u << 16);          // dim1 hi=0 | tile_dim0=768
    g1[4] = 32;                         // tile_dim1=32, tile_dim2=0
    g1[5] = 768;                        // tensor_dim0_stride lo32
    g1[6] = 0;                          // stride hi | dim1_stride lo
    g1[7] = 0;
#if defined(__clang_major__) && __clang_major__ >= 23
    __builtin_amdgcn_tensor_load_to_lds(g0, g1, (i32x4)0, (i32x4)0, (i32x8)0, 0);
#else
    __builtin_amdgcn_tensor_load_to_lds(g0, g1, (i32x4)0, (i32x4)0, 0);
#endif
    __builtin_amdgcn_s_wait_tensorcnt(0);
  }
#else
  {
    const unsigned* src = (const unsigned*)(xproj + (brow0 + qb * 32) * DIM);
    unsigned* dst = (unsigned*)s_Q;
    for (int i = tid; i < 32 * DIM / 2; i += 256) dst[i] = src[i];
  }
#endif
  if (tid < 32) { s_max[tid] = -__builtin_inff(); s_sum[tid] = 0.f; }
  __syncthreads();

  v8f Oacc[2][6];
#pragma unroll
  for (int mt = 0; mt < 2; ++mt)
#pragma unroll
    for (int nt = 0; nt < 6; ++nt) Oacc[mt][nt] = {};

  // S-matmul job split: 4 tiles x 2 K-halves over 8 waves
  const int mt_s  = (wave >> 1) & 1;
  const int ntk_s = wave & 1;
  const int kh_s  = wave >> 2;
  float* const s_Sh = kh_s ? s_S1 : s_S0;

  for (int jb = 0; jb < 32; ++jb) {
    if (tid < 32) s_mask[tid] = mask[brow0 + jb * 32 + tid];

    // partial S = Qp * Kp^T over this wave's K-half (12 WMMAs)
    {
      v8f c = {};
      const int qrow = 16 * mt_s + l15;
      const unsigned short* kr =
          xproj + (brow0 + jb * 32 + 16 * ntk_s + l15) * DIM;
      if (jb < 31)  // prefetch next K block row (-> global_prefetch_b8)
        __builtin_prefetch(xproj + (brow0 + (jb + 1) * 32 + 16 * ntk_s + l15) * DIM, 0, 1);
      const int kb0 = kh_s * (DIM / 2);
      for (int kb = kb0; kb < kb0 + DIM / 2; kb += 32) {
        ABReg a, bm;
#pragma unroll
        for (int v = 0; v < 8; ++v) {
          int k = kb + kpat(v, half);
          a.u[v]  = *(const unsigned*)&s_Q[qrow * DIM + k];
          bm.u[v] = *(const unsigned*)&kr[k];
        }
        c = __builtin_amdgcn_wmma_f32_16x16x32_bf16(false, a.v, false, bm.v,
                                                    (short)0, c, false, false);
      }
#pragma unroll
      for (int r = 0; r < 8; ++r) {
        int ml = 16 * mt_s + r + 8 * half;
        int nl = 16 * ntk_s + l15;
        s_Sh[ml * 32 + nl] = c[r];             // plain ds_store, disjoint bufs
      }
    }
    __syncthreads();

    // online softmax: one thread per query row
    if (tid < 32) {
      const int m = tid;
      const int iglob = qb * 32 + m;
      const float NEG_INF = -__builtin_inff();
      float mold = s_max[m];
      float rmax = mold;
      float sv[32];
#pragma unroll
      for (int j = 0; j < 32; ++j) {
        float s = s_S0[m * 32 + j] + s_S1[m * 32 + j];
        if (jb * 32 + j == iglob) s = 0.f;     // zero diagonal first
        if (s_mask[j])            s = NEG_INF; // then mask padded keys
        sv[j] = s;
        rmax = fmaxf(rmax, s);
      }
      float scale = 1.f;
      float rsum  = s_sum[m];
      if (rmax != NEG_INF) {
        scale = __expf(mold - rmax);           // mold==-inf -> 0, O is 0 anyway
        float acc = 0.f;
#pragma unroll
        for (int j = 0; j < 32; j += 2) {
          float p0 = (sv[j]     == NEG_INF) ? 0.f : __expf(sv[j]     - rmax);
          float p1 = (sv[j + 1] == NEG_INF) ? 0.f : __expf(sv[j + 1] - rmax);
          acc += p0 + p1;
          *(unsigned*)&s_P[m * 32 + j] = pack2(p0, p1);
        }
        rsum = rsum * scale + acc;
      } else {
#pragma unroll
        for (int j = 0; j < 32; j += 2) *(unsigned*)&s_P[m * 32 + j] = 0;
      }
      s_scale[m] = scale;
      s_max[m]   = rmax;
      s_sum[m]   = rsum;
    }
    __syncthreads();

    // rescale O accumulators by row scale
#pragma unroll
    for (int mt = 0; mt < 2; ++mt)
#pragma unroll
      for (int r = 0; r < 8; ++r) {
        float sc = s_scale[16 * mt + r + 8 * half];
#pragma unroll
        for (int nt = 0; nt < 6; ++nt) Oacc[mt][nt][r] *= sc;
      }

    // O += P @ V ; V from transposed bf16 xT -> pure dword loads
    for (int nt = 0; nt < 6; ++nt) {
      ABReg bv;
      const int ncol = (wave * 6 + nt) * 16 + l15;
      const unsigned short* vrow = xT + ((size_t)b * DIM + ncol) * SEQL + jb * 32;
#pragma unroll
      for (int v = 0; v < 8; ++v)
        bv.u[v] = *(const unsigned*)&vrow[kpat(v, half)];
#pragma unroll
      for (int mt = 0; mt < 2; ++mt) {
        ABReg a;
        const int prow = 16 * mt + l15;
#pragma unroll
        for (int v = 0; v < 8; ++v)
          a.u[v] = *(const unsigned*)&s_P[prow * 32 + kpat(v, half)];
        Oacc[mt][nt] = __builtin_amdgcn_wmma_f32_16x16x32_bf16(
            false, a.v, false, bv.v, (short)0, Oacc[mt][nt], false, false);
      }
    }
    __syncthreads();
  }

  // epilogue: divide by softmax denominator, store fp32
#pragma unroll
  for (int mt = 0; mt < 2; ++mt)
#pragma unroll
    for (int r = 0; r < 8; ++r) {
      int m = 16 * mt + r + 8 * half;
      float inv = 1.f / s_sum[m];
#pragma unroll
      for (int nt = 0; nt < 6; ++nt) {
        int ncol = (wave * 6 + nt) * 16 + l15;
        out[(brow0 + qb * 32 + m) * DIM + ncol] = Oacc[mt][nt][r] * inv;
      }
    }
}

// ---------------------------------------------------------------------------
extern "C" void kernel_launch(void* const* d_in, const int* in_sizes, int n_in,
                              void* d_out, int out_size, void* d_ws, size_t ws_size,
                              hipStream_t stream) {
  (void)in_sizes; (void)n_in; (void)out_size; (void)ws_size;
  const float*         x    = (const float*)d_in[0];
  const unsigned char* mask = (const unsigned char*)d_in[1]; // jnp.bool_ = 1 byte
  const float*         W    = (const float*)d_in[2];
  const float*         bias = (const float*)d_in[3];
  float*               out  = (float*)d_out;

  // workspace layout (bf16): xproj [B*L, D] | xT [B, D, L] | Wbf [D, D]
  const size_t nXP = (size_t)BATCH * SEQL * DIM;
  unsigned short* xproj = (unsigned short*)d_ws;
  unsigned short* xT    = xproj + nXP;
  unsigned short* Wbf   = xT + nXP;

  convw_kernel<<<(DIM * DIM / 2) / 256, 256, 0, stream>>>(W, Wbf);
  transx_kernel<<<BATCH * (SEQL / 64) * (DIM / 64), 256, 0, stream>>>(x, xT);
  proj_kernel<<<(1024 * 12) / 8, 256, 0, stream>>>(x, Wbf, bias, xproj);
  attn_kernel<<<BATCH * (SEQL / 32), 256, 0, stream>>>(xT, mask, xproj, out);
}
